// Net_18734647345154
// MI455X (gfx1250) — matrix operand order repaired and checked by reading, not compiled, not run on hardware
//
#include <hip/hip_runtime.h>

// MI455X / gfx1250: wave32, WMMA 16x16x32 bf16 (f32 accum).
// D=128 everywhere; one 256-thread block = 8 waves; each wave owns a
// 16-row x 128-col output slab = 8 accumulators of v8f (16x16 C tiles).
// Weights are staged into LDS with global_load_async_to_lds_b128 (ASYNCcnt),
// GroupNorm is done in-register via __shfl_xor over the WMMA C layout.

constexpr int kD   = 128;
constexpr int kN   = 16384;           // N_AGT == N_CTX
constexpr int kE   = 524288;
constexpr int kNND = kN * kD;         // 2,097,152 floats = 8 MB

typedef __attribute__((ext_vector_type(16))) __bf16 v16bf;
typedef __attribute__((ext_vector_type(8)))  float  v8f;

__device__ __forceinline__ unsigned short f2bfu(float f) {
  unsigned u = __float_as_uint(f);
  unsigned r = u + 0x7FFFu + ((u >> 16) & 1u);   // round-to-nearest-even
  return (unsigned short)(r >> 16);
}

// WMMA A-operand LDS image for a 128x128 bf16 tile.
// 16-bit A 16x32 (MxK): lane m (m<16) holds row m, K={kb+0..7, kb+16..23};
// lane m+16 holds row m, K={kb+8..15, kb+24..31}. Region per (wave,kt)=512 bf16.
__device__ __forceinline__ int a_index(int e, int k) {
  int w = e >> 4, m = e & 15;
  int kt = k >> 5, kk = k & 31;
  int lane = m + ((kk & 8) ? 16 : 0);
  int slot = (kk & 7) + ((kk & 16) ? 8 : 0);
  return ((w * 4 + kt) << 9) + (lane << 4) + slot;
}

__device__ __forceinline__ v16bf ld16(const unsigned short* p) {
  return *reinterpret_cast<const v16bf*>(p);
}

// Async 32KB weight image -> LDS: 2048 x b128, 8 per thread, GVS addressing
// (SGPR pair base + 32-bit VGPR offset; LDS byte address = low 32 bits of the
// generic shared pointer). Tracked with ASYNCcnt.
__device__ __forceinline__ void async_copy_w(const unsigned short* __restrict__ src,
                                             unsigned short* dst_lds, int tid) {
  unsigned lbase = (unsigned)(unsigned long long)dst_lds;
#pragma unroll
  for (int k = 0; k < 8; ++k) {
    unsigned goff  = (unsigned)((tid + (k << 8)) << 4);
    unsigned laddr = lbase + goff;
    asm volatile("global_load_async_to_lds_b128 %0, %1, %2"
                 :: "v"(laddr), "v"(goff), "s"(src)
                 : "memory");
  }
}
__device__ __forceinline__ void wait_async0() {
  asm volatile("s_wait_asynccnt 0" ::: "memory");
}

// 128x128 @ 128x128 per block: 4 K-tiles x 8 N-tiles per wave -> 32 v_wmma.
// A/B tile ds_loads are software-pipelined one step ahead to hide LDS latency
// behind the WMMA pipe (avoids s_wait_dscnt 0 before every issue).
__device__ __forceinline__ void wmma_tile_gemm(const unsigned short* lA,
                                               const unsigned short* lB,
                                               v8f acc[8], int wv, int lane) {
  const unsigned short* Ab = lA + (wv << 11) + (lane << 4);
  const unsigned short* Bb = lB + (lane << 4);
  v16bf a = ld16(Ab);
  v16bf b = ld16(Bb);
#pragma unroll
  for (int kt = 0; kt < 4; ++kt) {
    v16bf a_nxt = a;
    if (kt < 3) a_nxt = ld16(Ab + ((kt + 1) << 9));
#pragma unroll
    for (int nt = 0; nt < 8; ++nt) {
      int t = (kt << 3) + nt;
      v16bf b_nxt = b;
      if (t < 31) b_nxt = ld16(Bb + ((t + 1) << 9));
      acc[nt] = __builtin_amdgcn_wmma_f32_16x16x32_bf16(
          false, a, false, b, (short)0, acc[nt], false, false);
      b = b_nxt;
    }
    a = a_nxt;
  }
}

// Per-row GroupNorm stats straight from the C-layout accumulators:
// VGPR r of lanes 0-15 = row (base+r), lanes 16-31 = row (base+r+8).
// Each 16-lane half holds its row's 128 cols (8 tiles x 16 lanes).
__device__ __forceinline__ void row_stats(const v8f acc[8], float mu[8], float inv[8]) {
#pragma unroll
  for (int r = 0; r < 8; ++r) {
    float s = 0.f, ss = 0.f;
#pragma unroll
    for (int nt = 0; nt < 8; ++nt) { float v = acc[nt][r]; s += v; ss += v * v; }
#pragma unroll
    for (int m = 1; m < 16; m <<= 1) {       // reduce within each 16-lane half
      s  += __shfl_xor(s, m, 32);
      ss += __shfl_xor(ss, m, 32);
    }
    float mean = s * (1.0f / 128.0f);
    float var  = ss * (1.0f / 128.0f) - mean * mean;
    mu[r]  = mean;
    inv[r] = rsqrtf(var + 1e-5f);
  }
}

// Convert a [K=128, N=128] f32 weight (row-major, optional row offset) into the
// per-lane bf16 WMMA-B image: tile (kt,nt) of 512 bf16; lane l<16 -> col nt*16+l,
// K=kt*32+{0..7,16..23}; lane l>=16 -> col nt*16+(l-16), K=kt*32+{8..15,24..31}.
__global__ __launch_bounds__(256) void prep_weight_kernel(
    const float* __restrict__ W, int row_off, unsigned short* __restrict__ dst) {
  int idx = blockIdx.x * 256 + threadIdx.x;   // 0..16383
  int j  = idx & 15;
  int l  = (idx >> 4) & 31;
  int t  = idx >> 9;                          // kt*8 + nt
  int kt = t >> 3, nt = t & 7;
  int col = (nt << 4) + (l & 15);
  int k   = (kt << 5) + ((l >> 4) << 3) + (j & 7) + ((j >> 3) << 4);
  dst[idx] = f2bfu(W[(size_t)(row_off + k) * kD + col]);
}

__global__ void zero_kernel(float* __restrict__ p, int n) {
  int i = blockIdx.x * blockDim.x + threadIdx.x;
  if (i < n) p[i] = 0.f;
}

// Generic Y = epilogue(X @ W): optional GroupNorm(gamma,beta), residual, relu.
__global__ __launch_bounds__(256) void gemm128_gn_kernel(
    const float* __restrict__ X, const unsigned short* __restrict__ Wp,
    const float* __restrict__ gamma, const float* __restrict__ beta,
    const float* __restrict__ residual, int do_relu, float* __restrict__ Out) {
  __shared__ __align__(64) unsigned short lA[16384];
  __shared__ __align__(64) unsigned short lB[16384];
  const int tid = threadIdx.x, lane = tid & 31, wv = tid >> 5;
  const int r0 = blockIdx.x << 7;

  async_copy_w(Wp, lB, tid);   // weights stream in while we convert X

  { int e = tid >> 1; int c0 = (tid & 1) << 6;   // lA writes are wave-local
    const float4* xr = (const float4*)(X + (size_t)(r0 + e) * kD + c0);
#pragma unroll
    for (int i = 0; i < 16; ++i) {
      float4 x = xr[i];
      int c = c0 + (i << 2);
      lA[a_index(e, c + 0)] = f2bfu(x.x);
      lA[a_index(e, c + 1)] = f2bfu(x.y);
      lA[a_index(e, c + 2)] = f2bfu(x.z);
      lA[a_index(e, c + 3)] = f2bfu(x.w);
    }
  }
  wait_async0();
  __syncthreads();             // publish lB (and lA)

  const v8f vz = {0.f, 0.f, 0.f, 0.f, 0.f, 0.f, 0.f, 0.f};
  v8f acc[8];
#pragma unroll
  for (int nt = 0; nt < 8; ++nt) acc[nt] = vz;
  wmma_tile_gemm(lA, lB, acc, wv, lane);

  const int cb = lane & 15, hb = (lane >> 4) << 3;
  float mu[8], inv[8], gg[8], bb[8];
  if (gamma) {
    row_stats(acc, mu, inv);
#pragma unroll
    for (int nt = 0; nt < 8; ++nt) {
      gg[nt] = gamma[(nt << 4) + cb];
      bb[nt] = beta[(nt << 4) + cb];
    }
  }
#pragma unroll
  for (int r = 0; r < 8; ++r) {
    int row = r0 + (wv << 4) + r + hb;
    float* orow = Out + (size_t)row * kD;
    const float* rrow = residual ? residual + (size_t)row * kD : nullptr;
#pragma unroll
    for (int nt = 0; nt < 8; ++nt) {
      int col = (nt << 4) + cb;
      float v = acc[nt][r];
      if (gamma) v = (v - mu[r]) * inv[r] * gg[nt] + bb[nt];
      if (rrow) v += rrow[col];
      if (do_relu) v = fmaxf(v, 0.f);
      orow[col] = v;
    }
  }
}

// Fused per-edge pipeline: pos->dist layer (VALU) -> WMMA GEMM (dist_w2) ->
// GN+relu (in-register) -> WMMA GEMM (ctx_w1 rows 0..127) -> + qW[hi] + cW[wi]
// -> GN+relu -> atomic scatter-add into S[hi]. 128 edges per block.
// Both weight images are async-loaded into separate LDS buffers up front, so
// only two barriers are needed (lA rewrites are wave-local).
__global__ __launch_bounds__(256) void lanegcn_edge_kernel(
    const float* __restrict__ agt_ctrs, const float* __restrict__ ctx_ctrs,
    const int* __restrict__ hi, const int* __restrict__ wi,
    const float* __restrict__ dw1, const float* __restrict__ db1,
    const unsigned short* __restrict__ dw2p,
    const float* __restrict__ dg2, const float* __restrict__ db2,
    const unsigned short* __restrict__ w1ap,
    const float* __restrict__ cg1, const float* __restrict__ cb1,
    const float* __restrict__ qW, const float* __restrict__ cW,
    float* __restrict__ S) {
  __shared__ __align__(64) unsigned short lA[16384];
  __shared__ __align__(64) unsigned short lB1[16384];
  __shared__ __align__(64) unsigned short lB2[16384];
  __shared__ int   sHi[128];
  __shared__ int   sWi[128];
  __shared__ float sPx[128];
  __shared__ float sPy[128];

  const int tid = threadIdx.x, lane = tid & 31, wv = tid >> 5;
  const int e0 = blockIdx.x << 7;

  async_copy_w(dw2p, lB1, tid);    // GEMM1 weights
  async_copy_w(w1ap, lB2, tid);    // GEMM2 weights (consumed much later)

  if (tid < 128) {
    int h = hi[e0 + tid], w = wi[e0 + tid];
    sHi[tid] = h; sWi[tid] = w;
    sPx[tid] = agt_ctrs[2 * h]     - ctx_ctrs[2 * w];
    sPy[tid] = agt_ctrs[2 * h + 1] - ctx_ctrs[2 * w + 1];
    // Warm L2/L0 for the epilogue gathers (512B rows -> 2 lines each).
    const char* q = (const char*)(qW + (size_t)h * kD);
    const char* c = (const char*)(cW + (size_t)w * kD);
    __builtin_prefetch(q, 0, 1);  __builtin_prefetch(q + 256, 0, 1);
    __builtin_prefetch(c, 0, 1);  __builtin_prefetch(c + 256, 0, 1);
  }
  __syncthreads();                 // publish sPx/sPy/sHi/sWi

  // h1 = relu(pos @ dist_w1 + dist_b1) -> bf16 A-image (wave-local rows)
  { int e = tid >> 1; int c0 = (tid & 1) << 6;
    float px = sPx[e], py = sPy[e];
#pragma unroll 8
    for (int c = c0; c < c0 + 64; ++c) {
      float v = fmaf(px, dw1[c], fmaf(py, dw1[kD + c], db1[c]));
      lA[a_index(e, c)] = f2bfu(fmaxf(v, 0.f));
    }
  }
  wait_async0();
  __syncthreads();                 // publish lB1/lB2 (and lA)

  const v8f vz = {0.f, 0.f, 0.f, 0.f, 0.f, 0.f, 0.f, 0.f};
  v8f acc[8];
#pragma unroll
  for (int nt = 0; nt < 8; ++nt) acc[nt] = vz;
  wmma_tile_gemm(lA, lB1, acc, wv, lane);         // t = h1 @ dist_w2

  const int cb = lane & 15, hb = (lane >> 4) << 3;
  {
    float mu[8], inv[8], gg[8], bb[8];
    row_stats(acc, mu, inv);
#pragma unroll
    for (int nt = 0; nt < 8; ++nt) {
      gg[nt] = dg2[(nt << 4) + cb];
      bb[nt] = db2[(nt << 4) + cb];
    }
#pragma unroll
    for (int r = 0; r < 8; ++r) {
      int e = (wv << 4) + r + hb;                 // wave-local rows only
#pragma unroll
      for (int nt = 0; nt < 8; ++nt) {
        float v = (acc[nt][r] - mu[r]) * inv[r] * gg[nt] + bb[nt];
        lA[a_index(e, (nt << 4) + cb)] = f2bfu(fmaxf(v, 0.f));
      }
    }
  }
  // No barrier: lA rewrite is wave-local and per-wave LDS ops are in-order;
  // lB2 visibility was covered by the barrier above.
#pragma unroll
  for (int nt = 0; nt < 8; ++nt) acc[nt] = vz;
  wmma_tile_gemm(lA, lB2, acc, wv, lane);         // u = d @ ctx_w1[0:D]

  {
    // c1 = u + qW[hi] + cW[wi]  (gathers hit L2: 8MB arrays, prefetched)
#pragma unroll
    for (int r = 0; r < 8; ++r) {
      int e = (wv << 4) + r + hb;
      const float* qrow = qW + (size_t)sHi[e] * kD;
      const float* crow = cW + (size_t)sWi[e] * kD;
#pragma unroll
      for (int nt = 0; nt < 8; ++nt) {
        int col = (nt << 4) + cb;
        acc[nt][r] += qrow[col] + crow[col];
      }
    }
    float mu[8], inv[8], gg[8], bb[8];
    row_stats(acc, mu, inv);
#pragma unroll
    for (int nt = 0; nt < 8; ++nt) {
      gg[nt] = cg1[(nt << 4) + cb];
      bb[nt] = cb1[(nt << 4) + cb];
    }
#pragma unroll
    for (int r = 0; r < 8; ++r) {
      int e = (wv << 4) + r + hb;
      float* srow = S + (size_t)sHi[e] * kD;
#pragma unroll
      for (int nt = 0; nt < 8; ++nt) {
        float v = (acc[nt][r] - mu[r]) * inv[r] * gg[nt] + bb[nt];
        atomicAdd(&srow[(nt << 4) + cb], fmaxf(v, 0.f));   // S lives in L2
      }
    }
  }
}

// t = relu(GroupNorm(X)) row-wise, no GEMM. 128 rows/block, 2 threads/row.
__global__ __launch_bounds__(256) void rownorm_relu_kernel(
    const float* __restrict__ X, const float* __restrict__ g,
    const float* __restrict__ b, float* __restrict__ Out) {
  __shared__ float ps[256];
  __shared__ float pss[256];
  int tid = threadIdx.x; int e = tid >> 1; int half = tid & 1;
  size_t row = (size_t)blockIdx.x * 128 + e;
  const float4* xr = (const float4*)(X + row * kD + (half << 6));
  float4 v[16];
  float s = 0.f, ss = 0.f;
#pragma unroll
  for (int i = 0; i < 16; ++i) {
    float4 x = xr[i]; v[i] = x;
    s  += x.x + x.y + x.z + x.w;
    ss += x.x * x.x + x.y * x.y + x.z * x.z + x.w * x.w;
  }
  ps[tid] = s; pss[tid] = ss;
  __syncthreads();
  float S1 = ps[e * 2] + ps[e * 2 + 1];
  float S2 = pss[e * 2] + pss[e * 2 + 1];
  float mean = S1 * (1.0f / 128.0f);
  float var  = S2 * (1.0f / 128.0f) - mean * mean;
  float inv  = rsqrtf(var + 1e-5f);
  float4* outr = (float4*)(Out + row * kD + (half << 6));
  const float4* gr = (const float4*)(g + (half << 6));
  const float4* br = (const float4*)(b + (half << 6));
#pragma unroll
  for (int i = 0; i < 16; ++i) {
    float4 gv = gr[i], bv = br[i], x = v[i], o;
    o.x = fmaxf((x.x - mean) * inv * gv.x + bv.x, 0.f);
    o.y = fmaxf((x.y - mean) * inv * gv.y + bv.y, 0.f);
    o.z = fmaxf((x.z - mean) * inv * gv.z + bv.z, 0.f);
    o.w = fmaxf((x.w - mean) * inv * gv.w + bv.w, 0.f);
    outr[i] = o;
  }
}

extern "C" void kernel_launch(void* const* d_in, const int* in_sizes, int n_in,
                              void* d_out, int out_size, void* d_ws, size_t ws_size,
                              hipStream_t stream) {
  (void)in_sizes; (void)n_in; (void)out_size; (void)ws_size;
  const float* agts     = (const float*)d_in[0];
  const float* ctx      = (const float*)d_in[1];
  const float* agt_ctrs = (const float*)d_in[2];
  const float* ctx_ctrs = (const float*)d_in[3];
  const int*   hi       = (const int*)d_in[4];
  const int*   wi       = (const int*)d_in[5];
  const float* dist_w1  = (const float*)d_in[6];
  const float* dist_b1  = (const float*)d_in[7];
  const float* dist_w2  = (const float*)d_in[8];
  const float* dist_g2  = (const float*)d_in[9];
  const float* dist_b2  = (const float*)d_in[10];
  const float* query_w  = (const float*)d_in[11];
  const float* query_g  = (const float*)d_in[12];
  const float* query_b  = (const float*)d_in[13];
  const float* ctx_w1   = (const float*)d_in[14];
  const float* ctx_g1   = (const float*)d_in[15];
  const float* ctx_b1   = (const float*)d_in[16];
  const float* ctx_w2   = (const float*)d_in[17];
  const float* agt_w    = (const float*)d_in[18];
  const float* norm_g   = (const float*)d_in[19];
  const float* norm_b   = (const float*)d_in[20];
  const float* lin_w    = (const float*)d_in[21];
  const float* lin_g    = (const float*)d_in[22];
  const float* lin_b    = (const float*)d_in[23];
  float* out = (float*)d_out;

  // Workspace: 8 x 32KB prepped bf16 weights, then 5 x 8MB f32 node buffers.
  unsigned short* wp = (unsigned short*)d_ws;
  unsigned short* p_dw2 = wp + 0 * 16384;
  unsigned short* p_w1a = wp + 1 * 16384;
  unsigned short* p_w1b = wp + 2 * 16384;
  unsigned short* p_w1c = wp + 3 * 16384;
  unsigned short* p_qw  = wp + 4 * 16384;
  unsigned short* p_aw  = wp + 5 * 16384;
  unsigned short* p_cw2 = wp + 6 * 16384;
  unsigned short* p_lw  = wp + 7 * 16384;
  float* fb = (float*)(wp + 8 * 16384);
  float* B0 = fb;             // qn, later t
  float* B1 = B0 + kNND;      // qW
  float* B2 = B1 + kNND;      // cW, later a
  float* B3 = B2 + kNND;      // aW
  float* B4 = B3 + kNND;      // S (scatter accumulator)

  dim3 b256(256);
  prep_weight_kernel<<<64, b256, 0, stream>>>(dist_w2, 0,   p_dw2);
  prep_weight_kernel<<<64, b256, 0, stream>>>(ctx_w1,  0,   p_w1a);
  prep_weight_kernel<<<64, b256, 0, stream>>>(ctx_w1,  128, p_w1b);
  prep_weight_kernel<<<64, b256, 0, stream>>>(ctx_w1,  256, p_w1c);
  prep_weight_kernel<<<64, b256, 0, stream>>>(query_w, 0,   p_qw);
  prep_weight_kernel<<<64, b256, 0, stream>>>(agt_w,   0,   p_aw);
  prep_weight_kernel<<<64, b256, 0, stream>>>(ctx_w2,  0,   p_cw2);
  prep_weight_kernel<<<64, b256, 0, stream>>>(lin_w,   0,   p_lw);

  zero_kernel<<<kNND / 256, b256, 0, stream>>>(B4, kNND);

  // Per-node precompute (gathers commute with row-wise GN / linear maps):
  gemm128_gn_kernel<<<128, b256, 0, stream>>>(agts, p_qw, query_g, query_b, nullptr, 1, B0); // qn
  gemm128_gn_kernel<<<128, b256, 0, stream>>>(B0, p_w1b, nullptr, nullptr, nullptr, 0, B1);  // qW = qn@W1b
  gemm128_gn_kernel<<<128, b256, 0, stream>>>(ctx, p_w1c, nullptr, nullptr, nullptr, 0, B2); // cW = ctx@W1c
  gemm128_gn_kernel<<<128, b256, 0, stream>>>(agts, p_aw, nullptr, nullptr, nullptr, 0, B3); // aW = agts@agt_w

  // Fused edge pipeline (2 WMMA GEMMs + 2 GNs + scatter), 128 edges/block:
  lanegcn_edge_kernel<<<kE / 128, b256, 0, stream>>>(
      agt_ctrs, ctx_ctrs, hi, wi, dist_w1, dist_b1, p_dw2, dist_g2, dist_b2,
      p_w1a, ctx_g1, ctx_b1, B1, B2, B4);

  // a = S @ ctx_w2 + aW   (ctx_w2 folded past the linear scatter-add)
  gemm128_gn_kernel<<<128, b256, 0, stream>>>(B4, p_cw2, nullptr, nullptr, B3, 0, B2);
  // t = relu(GN(a))
  rownorm_relu_kernel<<<128, b256, 0, stream>>>(B2, norm_g, norm_b, B0);
  // out = relu(GN(t @ lin_w) + agts)
  gemm128_gn_kernel<<<128, b256, 0, stream>>>(B0, p_lw, lin_g, lin_b, agts, 1, out);
}
